// Windowattn_block_20504173871321
// MI455X (gfx1250) — matrix operand order repaired and checked by reading, not compiled, not run on hardware
//
#include <hip/hip_runtime.h>
#include <math.h>

// ---------------- feature probes (gfx1250 async / TDM paths) ----------------
#if defined(__has_builtin)
# if __has_builtin(__builtin_amdgcn_tensor_load_to_lds) && __has_builtin(__builtin_amdgcn_s_wait_tensorcnt)
#  define USE_TDM 1
# endif
# if __has_builtin(__builtin_amdgcn_global_load_async_to_lds_b128) && __has_builtin(__builtin_amdgcn_s_wait_asynccnt)
#  define USE_ASYNC 1
# endif
#endif

// ---------------- types / helpers ----------------
typedef __attribute__((ext_vector_type(16))) __bf16 bf16x16;
typedef __attribute__((ext_vector_type(8)))  float  f32x8;
typedef __attribute__((ext_vector_type(4)))  unsigned u32x4;
typedef __attribute__((ext_vector_type(8)))  int      i32x8;
typedef __attribute__((ext_vector_type(4)))  int      i32x4;

struct alignas(16) U4 { unsigned a, b, c, d; };

union FragB {
    bf16x16        v;
    U4             q[2];
    unsigned short s[16];
};

__device__ __forceinline__ unsigned short f2bf(float f) {
    unsigned u = __builtin_bit_cast(unsigned, f);
    unsigned r = u + 0x7FFFu + ((u >> 16) & 1u);
    return (unsigned short)(r >> 16);
}

#ifdef USE_TDM
__device__ __forceinline__ unsigned lds_byte_off(const void* p) {
    return (unsigned)(size_t)(__attribute__((address_space(3))) const void*)p;
}

// 2D bf16 tile load via Tensor Data Mover, with HW LDS padding:
// rows of tile0=32 elems (64B = 16 dwords), pad 4 dwords (8 elems) -> LDS stride 40 elems.
__device__ __forceinline__ void tdm_load_2d(const void* gptr, unsigned ldsOff,
                                            unsigned dim0, unsigned dim1,
                                            unsigned tile0, unsigned tile1,
                                            unsigned stride0) {
    unsigned long long ga = (unsigned long long)(size_t)gptr;
    u32x4 g0;
    g0[0] = 1u;                                            // count=1 (valid descriptor)
    g0[1] = ldsOff;                                        // lds_addr
    g0[2] = (unsigned)(ga & 0xFFFFFFFFu);                  // global_addr[31:0]
    g0[3] = (unsigned)((ga >> 32) & 0x01FFFFFFu) | (2u << 30); // addr[56:32] | type=2
    i32x8 g1;
    g1[0] = (int)((1u << 16) | (1u << 20) | (3u << 22) | (3u << 25)); // 2B elems, pad_en, every 16dw pad 4dw
    g1[1] = (int)((dim0 & 0xFFFFu) << 16);                 // tensor_dim0[15:0]
    g1[2] = (int)((dim0 >> 16) | ((dim1 & 0xFFFFu) << 16));// tensor_dim0[31:16], tensor_dim1[15:0]
    g1[3] = (int)((dim1 >> 16) | (tile0 << 16));           // tensor_dim1[31:16], tile_dim0
    g1[4] = (int)(tile1 & 0xFFFFu);                        // tile_dim1 (tile_dim2 = 0)
    g1[5] = (int)stride0;                                  // tensor_dim0_stride[31:0]
    g1[6] = 0;                                             // stride0[47:32], dim1_stride[15:0]
    g1[7] = 0;
    i32x4 gz4 = {0, 0, 0, 0};
    i32x8 gz8 = {0, 0, 0, 0, 0, 0, 0, 0};
    __builtin_amdgcn_tensor_load_to_lds(g0, g1, gz4, gz4, gz8, 0);
}
#endif

#ifdef USE_ASYNC
__device__ __forceinline__ void async_cp16(unsigned short* lds, const unsigned short* g) {
    __builtin_amdgcn_global_load_async_to_lds_b128(
        (__attribute__((address_space(1))) i32x4*)g,
        (__attribute__((address_space(3))) i32x4*)lds, 0, 0);
}
#endif

// geometry: B=1, Z=8, H=48, W=96, C=512; windows (2,6,12) -> 256 windows of N=144
#define NTOK   36864

// ---------------- layernorm (+bf16 convert, optional window permute) ----------------
__global__ __launch_bounds__(256) void ln_kernel(const float* __restrict__ X,
                                                 const float* __restrict__ w,
                                                 const float* __restrict__ b,
                                                 unsigned short* __restrict__ Out,
                                                 int remap) {
    __shared__ float red[256];
    const int t = blockIdx.x, tid = threadIdx.x;
    float v0 = X[(size_t)t * 512 + tid];
    float v1 = X[(size_t)t * 512 + 256 + tid];
    red[tid] = v0 + v1;
    __syncthreads();
    for (int o = 128; o > 0; o >>= 1) { if (tid < o) red[tid] += red[tid + o]; __syncthreads(); }
    float mu = red[0] * (1.0f / 512.0f);
    __syncthreads();
    float d0 = v0 - mu, d1 = v1 - mu;
    red[tid] = d0 * d0 + d1 * d1;
    __syncthreads();
    for (int o = 128; o > 0; o >>= 1) { if (tid < o) red[tid] += red[tid + o]; __syncthreads(); }
    float rstd = rsqrtf(red[0] * (1.0f / 512.0f) + 1e-5f);
    int m = t;
    if (remap) {  // token (z,h,w) -> window-major row
        int z = t / 4608, rem = t % 4608, hh = rem / 96, wd = rem % 96;
        int zb = z >> 1, iz = z & 1, hb = hh / 6, ih = hh % 6, wb = wd / 12, iw = wd % 12;
        m = ((zb * 8 + hb) * 8 + wb) * 144 + (iz * 6 + ih) * 12 + iw;
    }
    Out[(size_t)m * 512 + tid]       = f2bf(d0 * rstd * w[tid] + b[tid]);
    Out[(size_t)m * 512 + 256 + tid] = f2bf(d1 * rstd * w[tid + 256] + b[tid + 256]);
}

// ---------------- fp32 -> bf16 weight convert ----------------
__global__ void cvt_kernel(const float* __restrict__ in, unsigned short* __restrict__ out, int n) {
    int i = blockIdx.x * 256 + threadIdx.x;
    if (i < n) out[i] = f2bf(in[i]);
}

// ---------------- generic WMMA bf16 GEMM: Out = A[M,K] * W[N,K]^T (+epilogue) ----------------
// MODE 0: +bias -> bf16             (QKV)
// MODE 1: +bias, exact GELU -> bf16 (FC1)
// MODE 2: +bias, window->token unpermute, + x residual -> f32 (proj -> x1)
// MODE 3: +bias, + x1 residual -> f32 (FC2 -> d_out)
template <int MODE>
__global__ __launch_bounds__(256) void gemm_bf16(const unsigned short* __restrict__ A,
                                                 const unsigned short* __restrict__ W,
                                                 const float* __restrict__ bias,
                                                 const float* __restrict__ resid,
                                                 void* __restrict__ Out,
                                                 int M, int N, int K) {
    __shared__ __align__(16) unsigned short As[128 * 40];   // stride 40 avoids bank conflicts
#ifdef USE_TDM
    __shared__ __align__(16) unsigned short Bs[128 * 40];   // W tile as-is: [N=128][K=32] stride 40
#else
    __shared__ __align__(16) unsigned short Bs[32 * 136];   // W tile transposed: [K=32][N=128+pad]
#endif
    const int tid = threadIdx.x;
    const int lane = tid & 31, wid = tid >> 5;
    const int wm = wid & 3, wn = wid >> 2;                  // 4x2 wave grid -> 32x64 each
    const int m0 = blockIdx.x * 128, n0 = blockIdx.y * 128;
    const int half = lane >> 4, l15 = lane & 15, kb = half * 8;

    f32x8 acc[2][4];
    const f32x8 zacc = {0.f, 0.f, 0.f, 0.f, 0.f, 0.f, 0.f, 0.f};
#pragma unroll
    for (int i = 0; i < 2; ++i)
#pragma unroll
        for (int j = 0; j < 4; ++j) acc[i][j] = zacc;

    for (int k0 = 0; k0 < K; k0 += 32) {
#ifdef USE_TDM
        if (tid < 32) {  // one wave drives the Tensor Data Mover for both tiles
            tdm_load_2d(A + (size_t)m0 * K + k0, lds_byte_off(As),
                        (unsigned)K, (unsigned)M, 32u, 128u, (unsigned)K);
            tdm_load_2d(W + (size_t)n0 * K + k0, lds_byte_off(Bs),
                        (unsigned)K, (unsigned)N, 32u, 128u, (unsigned)K);
            __builtin_amdgcn_s_wait_tensorcnt(0);
        }
        __syncthreads();
#else
#pragma unroll
        for (int i = 0; i < 2; ++i) {       // stage A: 128x32 bf16
            int chunk = tid * 2 + i, r = chunk >> 2, cid = chunk & 3;
            U4 dta = *(const U4*)(A + (size_t)(m0 + r) * K + k0 + cid * 8);
            *(U4*)(As + r * 40 + cid * 8) = dta;
        }
#pragma unroll
        for (int i = 0; i < 2; ++i) {       // stage W transposed: Bs[k][n]
            int chunk = tid * 2 + i, r = chunk >> 2, cid = chunk & 3;
            U4 dtw = *(const U4*)(W + (size_t)(n0 + r) * K + k0 + cid * 8);
            const unsigned short* e = (const unsigned short*)&dtw;
#pragma unroll
            for (int j = 0; j < 8; ++j) Bs[(cid * 8 + j) * 136 + r] = e[j];
        }
        if (k0 + 32 < K) {                  // gfx1250 global_prefetch_b8 of next tiles
            __builtin_prefetch(A + (size_t)(m0 + (tid >> 1)) * K + k0 + 32, 0, 3);
            __builtin_prefetch(W + (size_t)(n0 + (tid >> 1)) * K + k0 + 32, 0, 3);
        }
        __syncthreads();
#endif

        FragB afr[2];
#pragma unroll
        for (int mt = 0; mt < 2; ++mt) {    // A frag: lane&15 = row, K = kb..kb+7, kb+16..kb+23
            const unsigned short* p = As + (wm * 32 + mt * 16 + l15) * 40 + kb;
            afr[mt].q[0] = *(const U4*)(p);
            afr[mt].q[1] = *(const U4*)(p + 16);
        }
#pragma unroll
        for (int nt = 0; nt < 4; ++nt) {    // B frag: lane&15 = N, K = kb..kb+7, kb+16..kb+23
            FragB bfr;
            const int nbase = wn * 64 + nt * 16;
#ifdef USE_TDM
            const unsigned short* p = Bs + (nbase + l15) * 40 + kb;
            bfr.q[0] = *(const U4*)(p);
            bfr.q[1] = *(const U4*)(p + 16);
#else
#pragma unroll
            for (int j = 0; j < 16; ++j)
                bfr.s[j] = Bs[(kb + (j & 7) + ((j >> 3) << 4)) * 136 + nbase + l15];
#endif
#pragma unroll
            for (int mt = 0; mt < 2; ++mt)
                acc[mt][nt] = __builtin_amdgcn_wmma_f32_16x16x32_bf16(
                    false, afr[mt].v, false, bfr.v, (short)0, acc[mt][nt], false, false);
        }
        __syncthreads();
    }

#pragma unroll
    for (int mt = 0; mt < 2; ++mt)
#pragma unroll
        for (int nt = 0; nt < 4; ++nt)
#pragma unroll
            for (int g = 0; g < 8; ++g) {
                int row = m0 + wm * 32 + mt * 16 + half * 8 + g;
                int col = n0 + wn * 64 + nt * 16 + l15;
                float v = acc[mt][nt][g] + bias[col];
                if (MODE == 0) {
                    ((unsigned short*)Out)[(size_t)row * N + col] = f2bf(v);
                } else if (MODE == 1) {
                    float gel = 0.5f * v * (1.0f + erff(v * 0.70710678118654752f));
                    ((unsigned short*)Out)[(size_t)row * N + col] = f2bf(gel);
                } else if (MODE == 2) {
                    int wI = row / 144, nI = row % 144;
                    int zb = wI >> 6, hb = (wI >> 3) & 7, wb = wI & 7;
                    int iz = nI / 72, ih = (nI / 12) % 6, iw = nI % 12;
                    int t = ((zb * 2 + iz) * 48 + (hb * 6 + ih)) * 96 + (wb * 12 + iw);
                    ((float*)Out)[(size_t)t * N + col] = resid[(size_t)t * N + col] + v;
                } else {
                    ((float*)Out)[(size_t)row * N + col] = resid[(size_t)row * N + col] + v;
                }
            }
}

// ---------------- window attention: one wave per (window, head) ----------------
__global__ __launch_bounds__(32) void attn_kernel(const unsigned short* __restrict__ QKV, // [36864,1536] bf16
                                                  const float* __restrict__ rpb,          // [759,16]
                                                  const int* __restrict__ rpi,            // [144,144]
                                                  unsigned short* __restrict__ Outb) {    // [36864,512] bf16
    __shared__ __align__(16) unsigned short v_s[160 * 32];     // V, K-padded to 160
    __shared__ __align__(16) unsigned short strip[16 * 160];   // one 16-row P strip (bf16)
    const int w = blockIdx.x, h = blockIdx.y;
    const int lane = threadIdx.x;
    const int half = lane >> 4, l15 = lane & 15, kb = half * 8;

    const unsigned short* Qb = QKV + (size_t)w * 144 * 1536 + h * 32;
    const unsigned short* Kb = Qb + 512;
    const unsigned short* Vb = Qb + 1024;

#ifdef USE_ASYNC
    for (int n = lane; n < 144; n += 32)                        // async DMA of V into LDS
#pragma unroll
        for (int c = 0; c < 4; ++c)
            async_cp16(v_s + n * 32 + c * 8, Vb + (size_t)n * 1536 + c * 8);
#else
    for (int n = lane; n < 144; n += 32)
#pragma unroll
        for (int c = 0; c < 4; ++c)
            *(U4*)(v_s + n * 32 + c * 8) = *(const U4*)(Vb + (size_t)n * 1536 + c * 8);
#endif
    if (lane < 16) {
        const U4 z = {0u, 0u, 0u, 0u};
#pragma unroll
        for (int c = 0; c < 4; ++c) *(U4*)(v_s + (144 + lane) * 32 + c * 8) = z;
    }

    // K^T fragments: B[k][n=lane&15] = K[n][k], k = kb..kb+7, kb+16..kb+23  (overlaps with async V copy)
    FragB bq[9];
#pragma unroll
    for (int nt = 0; nt < 9; ++nt) {
        const unsigned short* pk = Kb + (size_t)(nt * 16 + l15) * 1536 + kb;
        bq[nt].q[0] = *(const U4*)(pk);
        bq[nt].q[1] = *(const U4*)(pk + 16);
    }

#ifdef USE_ASYNC
    __builtin_amdgcn_s_wait_asynccnt(0);
#endif
    __syncthreads();

    const float scale = 0.17677669529663687f;  // 1/sqrt(32)
    const f32x8 zacc = {0.f, 0.f, 0.f, 0.f, 0.f, 0.f, 0.f, 0.f};

    for (int mt = 0; mt < 9; ++mt) {
        // ---- S(mt,:) = Q(mt) * K^T : 9 WMMAs, K=32 in exactly one step ----
        FragB afr;
        const unsigned short* p = Qb + (size_t)(mt * 16 + l15) * 1536 + kb;
        afr.q[0] = *(const U4*)(p);
        afr.q[1] = *(const U4*)(p + 16);
        f32x8 acc[9];
#pragma unroll
        for (int nt = 0; nt < 9; ++nt)
            acc[nt] = __builtin_amdgcn_wmma_f32_16x16x32_bf16(
                false, afr.v, false, bq[nt].v, (short)0, zacc, false, false);

        // ---- exact softmax over the full 144-wide row strip (registers + shfl) ----
#pragma unroll
        for (int g = 0; g < 8; ++g) {
            int row = mt * 16 + half * 8 + g;
            float s[9];
            float mx = -3.0e38f;
#pragma unroll
            for (int nt = 0; nt < 9; ++nt) {
                int col = nt * 16 + l15;
                float bv = rpb[rpi[row * 144 + col] * 16 + h];
                s[nt] = acc[nt][g] * scale + bv;
                mx = fmaxf(mx, s[nt]);
            }
#pragma unroll
            for (int off = 8; off >= 1; off >>= 1) mx = fmaxf(mx, __shfl_xor(mx, off, 32));
            float sum = 0.f;
#pragma unroll
            for (int nt = 0; nt < 9; ++nt) { s[nt] = __expf(s[nt] - mx); sum += s[nt]; }
#pragma unroll
            for (int off = 8; off >= 1; off >>= 1) sum += __shfl_xor(sum, off, 32);
            float inv = 1.0f / sum;
            int sr = half * 8 + g;
#pragma unroll
            for (int nt = 0; nt < 9; ++nt) strip[sr * 160 + nt * 16 + l15] = f2bf(s[nt] * inv);
            strip[sr * 160 + 144 + l15] = 0;   // zero K-pad
        }
        __syncthreads();

        // ---- O(mt) = P(mt) * V : K = 160 in 5 steps, N = 32 in 2 tiles ----
        f32x8 o0 = zacc, o1 = zacc;
        for (int ks = 0; ks < 5; ++ks) {
            FragB a2;
            const unsigned short* pp = strip + l15 * 160 + ks * 32 + kb;
            a2.q[0] = *(const U4*)(pp);
            a2.q[1] = *(const U4*)(pp + 16);
            FragB b0, b1;
            const unsigned short* pv = v_s + ks * 32 * 32 + l15;  // B[k][n=lane&15], k = kb..kb+7, kb+16..kb+23
#pragma unroll
            for (int j = 0; j < 16; ++j) {
                int kk = kb + (j & 7) + ((j >> 3) << 4);
                b0.s[j] = pv[kk * 32];
                b1.s[j] = pv[kk * 32 + 16];
            }
            o0 = __builtin_amdgcn_wmma_f32_16x16x32_bf16(false, a2.v, false, b0.v, (short)0, o0, false, false);
            o1 = __builtin_amdgcn_wmma_f32_16x16x32_bf16(false, a2.v, false, b1.v, (short)0, o1, false, false);
        }
#pragma unroll
        for (int g = 0; g < 8; ++g) {
            int row = mt * 16 + half * 8 + g;
            size_t base = (size_t)(w * 144 + row) * 512 + h * 32;
            Outb[base + l15]      = f2bf(o0[g]);
            Outb[base + 16 + l15] = f2bf(o1[g]);
        }
        __syncthreads();   // strip reused next mt
    }
}

// ---------------- launch ----------------
extern "C" void kernel_launch(void* const* d_in, const int* in_sizes, int n_in,
                              void* d_out, int out_size, void* d_ws, size_t ws_size,
                              hipStream_t stream) {
    const float* x      = (const float*)d_in[0];
    const float* ln1_w  = (const float*)d_in[1];
    const float* ln1_b  = (const float*)d_in[2];
    const float* qkv_w  = (const float*)d_in[3];
    const float* qkv_b  = (const float*)d_in[4];
    const float* rpb    = (const float*)d_in[5];
    const float* proj_w = (const float*)d_in[6];
    const float* proj_b = (const float*)d_in[7];
    const float* ln2_w  = (const float*)d_in[8];
    const float* ln2_b  = (const float*)d_in[9];
    const float* fc1_w  = (const float*)d_in[10];
    const float* fc1_b  = (const float*)d_in[11];
    const float* fc2_w  = (const float*)d_in[12];
    const float* fc2_b  = (const float*)d_in[13];
    const int*   rpi    = (const int*)d_in[14];

    char* ws = (char*)d_ws;
    float*          x1   = (float*)(ws);                              // 75,497,472 B: x + attn (token order)
    unsigned short* bufA = (unsigned short*)(ws + 75497472);          // 37,748,736 B: ln1 -> attn_out -> ln2out
    unsigned short* bufB = (unsigned short*)(ws + 113246208);         // 150,994,944 B: qkv -> fc1out
    unsigned short* wq   = (unsigned short*)(ws + 264241152);
    unsigned short* wp   = (unsigned short*)(ws + 265814016);
    unsigned short* w1   = (unsigned short*)(ws + 266338304);
    unsigned short* w2   = (unsigned short*)(ws + 268435456);

    cvt_kernel<<<(1536 * 512 + 255) / 256, 256, 0, stream>>>(qkv_w,  wq, 1536 * 512);
    cvt_kernel<<<(512 * 512 + 255) / 256,  256, 0, stream>>>(proj_w, wp, 512 * 512);
    cvt_kernel<<<(2048 * 512 + 255) / 256, 256, 0, stream>>>(fc1_w,  w1, 2048 * 512);
    cvt_kernel<<<(512 * 2048 + 255) / 256, 256, 0, stream>>>(fc2_w,  w2, 512 * 2048);

    // LN1: token order -> window-major bf16
    ln_kernel<<<NTOK, 256, 0, stream>>>(x, ln1_w, ln1_b, bufA, 1);
    // QKV: [36864,512] x [1536,512]^T
    gemm_bf16<0><<<dim3(288, 12), 256, 0, stream>>>(bufA, wq, qkv_b, nullptr, bufB, NTOK, 1536, 512);
    // window attention (256 windows x 16 heads)
    attn_kernel<<<dim3(256, 16), 32, 0, stream>>>(bufB, rpb, rpi, bufA);
    // proj + un-permute + residual -> x1 (f32, token order)
    gemm_bf16<2><<<dim3(288, 4), 256, 0, stream>>>(bufA, wp, proj_b, x, x1, NTOK, 512, 512);
    // LN2 (token order)
    ln_kernel<<<NTOK, 256, 0, stream>>>(x1, ln2_w, ln2_b, bufA, 0);
    // FC1 + exact GELU
    gemm_bf16<1><<<dim3(288, 16), 256, 0, stream>>>(bufA, w1, fc1_b, nullptr, bufB, NTOK, 2048, 512);
    // FC2 + residual -> d_out (f32)
    gemm_bf16<3><<<dim3(288, 4), 256, 0, stream>>>(bufB, w2, fc2_b, x1, (float*)d_out, NTOK, 512, 2048);
}
